// MoSRAHLayer_49941879718136
// MI455X (gfx1250) — matrix operand order repaired and checked by reading, not compiled, not run on hardware
//
#include <hip/hip_runtime.h>
#include <hip/hip_bf16.h>

// Problem constants (fixed by the reference)
#define B_    4
#define N_    2048
#define D_    1024
#define E_    8
#define BTL_  128
#define M_OUT (B_ * N_ * D_)

typedef __attribute__((ext_vector_type(16))) __bf16 v16bf;
typedef __attribute__((ext_vector_type(8)))  __bf16 v8bf;
typedef __attribute__((ext_vector_type(8)))  float  v8f;

__device__ __forceinline__ v8f vzero8() {
  v8f z = {0.f, 0.f, 0.f, 0.f, 0.f, 0.f, 0.f, 0.f};
  return z;
}

__device__ __forceinline__ v8f wmma_bf16(v16bf a, v16bf b, v8f c) {
  return __builtin_amdgcn_wmma_f32_16x16x32_bf16(
      false, a, false, b, (short)0, c, false, false);
}

// A-fragment element mapping (16x32 bf16):
//   lane = m + 16*(k>>4) ; slot = k&15
// => per lane, slots 0..15 are k=16g..16g+15 (contiguous).
__device__ __forceinline__ int afrag_slot(int k) { return k & 15; }
__device__ __forceinline__ int afrag_lane(int m, int k) { return m + ((k >> 4) << 4); }

// B-fragment (32x16): per lane, slot i holds B(k = 16*g + i, n = lane&15).
// With a K-contiguous (transposed) source, a lane's fragment = 32 contiguous bytes.
__device__ __forceinline__ v16bf load_bf16_frag(const __bf16* p, int g) {
  const v8bf r0 = *(const v8bf*)(p + 16 * g);
  const v8bf r1 = *(const v8bf*)(p + 16 * g + 8);
  v16bf a;
#pragma unroll
  for (int i = 0; i < 8; ++i) { a[i] = r0[i]; a[8 + i] = r1[i]; }
  return a;
}

// ------------------------------------------------------------------ init
__global__ void k_init(float* __restrict__ out, float* __restrict__ red) {
  const long i = (long)blockIdx.x * blockDim.x + threadIdx.x;
  const long stride = (long)gridDim.x * blockDim.x;
  for (long j = i; j < (long)M_OUT; j += stride) out[j] = 0.f;
  if (i < 32) red[i] = 0.f;
}

// ------------------------------------------------------------------ prep: bf16 x + transposed bf16 weights
__global__ void k_prep(const float* __restrict__ x, const float* __restrict__ wd,
                       const float* __restrict__ wq, const float* __restrict__ wk,
                       const float* __restrict__ wv, const float* __restrict__ wup,
                       __bf16* __restrict__ xb,
                       __bf16* __restrict__ wdT, __bf16* __restrict__ wqT,
                       __bf16* __restrict__ wkT, __bf16* __restrict__ wvT,
                       __bf16* __restrict__ wupT) {
  const long i = (long)blockIdx.x * blockDim.x + threadIdx.x;
  const long stride = (long)gridDim.x * blockDim.x;
  for (long j = i; j < (long)B_ * N_ * D_; j += stride)          // 8M, coalesced
    xb[j] = (__bf16)x[j];
  for (long j = i; j < (long)E_ * BTL_ * D_; j += stride) {      // 1M
    const long e = j >> 17, c = (j >> 10) & 127, d = j & 1023;
    wdT[j] = (__bf16)wd[((e << 10) + d) * BTL_ + c];
  }
  for (long j = i; j < (long)E_ * BTL_ * BTL_; j += stride) {    // 128K
    const long e = j >> 14, c = (j >> 7) & 127, kk = j & 127;
    const long s = ((e << 7) + kk) * BTL_ + c;
    wqT[j] = (__bf16)wq[s];
    wkT[j] = (__bf16)wk[s];
    wvT[j] = (__bf16)wv[s];
  }
  for (long j = i; j < (long)E_ * D_ * BTL_; j += stride) {      // 1M
    const long e = j >> 17, c = (j >> 7) & 1023, kk = j & 127;
    wupT[j] = (__bf16)wup[((e << 7) + kk) * D_ + c];
  }
}

// ------------------------------------------------------------------ router
__global__ void k_router(const float* __restrict__ x, const unsigned char* __restrict__ act,
                         const float* __restrict__ wg, float* __restrict__ gatew,
                         float* __restrict__ red) {
  const int t = blockIdx.x * blockDim.x + threadIdx.x;
  if (t >= B_ * N_) return;
  float acc[E_];
#pragma unroll
  for (int e = 0; e < E_; ++e) acc[e] = 0.f;
  const float* xr = x + (long)t * D_;
  for (int d = 0; d < D_; ++d) {
    const float xv = xr[d];
#pragma unroll
    for (int e = 0; e < E_; ++e) acc[e] += xv * wg[d * E_ + e];
  }
  float mx = acc[0];
#pragma unroll
  for (int e = 1; e < E_; ++e) mx = fmaxf(mx, acc[e]);
  float p[E_], s = 0.f;
#pragma unroll
  for (int e = 0; e < E_; ++e) { p[e] = __expf(acc[e] - mx); s += p[e]; }
  const float inv = 1.f / s;
#pragma unroll
  for (int e = 0; e < E_; ++e) p[e] *= inv;
  int e1 = 0; float b1 = p[0];
#pragma unroll
  for (int e = 1; e < E_; ++e) if (p[e] > b1) { b1 = p[e]; e1 = e; }
  int e2 = (e1 == 0) ? 1 : 0; float b2 = p[e2];
#pragma unroll
  for (int e = 0; e < E_; ++e) if (e != e1 && p[e] > b2) { b2 = p[e]; e2 = e; }
  const float gs = b1 + b2;
  const float g1 = b1 / gs, g2 = b2 / gs;
  const float a = act[t] ? 1.f : 0.f;
#pragma unroll
  for (int e = 0; e < E_; ++e) {
    const float gw = (e == e1) ? g1 : ((e == e2) ? g2 : 0.f);
    gatew[(long)t * E_ + e] = gw * a;
  }
  if (act[t]) {
#pragma unroll
    for (int e = 0; e < E_; ++e) atomicAdd(&red[8 + e], p[e]);
    atomicAdd(&red[e1], 1.f);
    atomicAdd(&red[e2], 1.f);
    atomicAdd(&red[16], 1.f);
  }
}

// ------------------------------------------------------------------ loss
__global__ void k_loss(const float* __restrict__ red, const int* __restrict__ topk_p,
                       float* __restrict__ out) {
  if (threadIdx.x != 0 || blockIdx.x != 0) return;
  const float topk = (float)topk_p[0];
  const float denom = fmaxf(red[16], 1.f);
  float lbl = 0.f, mxf = 0.f;
  for (int e = 0; e < E_; ++e) {
    const float frac = red[e] / denom;
    const float pm = red[8 + e] / denom;
    lbl += frac * pm;
    mxf = fmaxf(mxf, frac);
  }
  lbl *= (float)E_ / topk;
  out[M_OUT] = lbl;
  out[M_OUT + 1] = mxf / (topk / (float)E_) - 1.f;
}

// ------------------------------------------------------------------ qkv
// h = x@W_down[e] (A-frags from bf16 x, B-frags from wdT, all direct global), h kept in
// LDS in A-fragment-native layout; q/k row-major bf16 to ws; v transposed (vT[b][c][tok]).
__global__ __launch_bounds__(256) void k_qkv(
    const __bf16* __restrict__ xb, const __bf16* __restrict__ wdT,
    const __bf16* __restrict__ wqT, const __bf16* __restrict__ wkT,
    const __bf16* __restrict__ wvT,
    __bf16* __restrict__ qo, __bf16* __restrict__ ko, __bf16* __restrict__ vT, int e) {
  __shared__ __align__(32) __bf16 hfrag[8 * 4 * 32 * 16];  // [wave][kc][lane][16] = 32 KB
  const int tid = threadIdx.x, lane = tid & 31, wave = tid >> 5;
  const int g = lane >> 4, nlo = lane & 15, m = lane & 15;
  const int b = blockIdx.y, row0 = blockIdx.x * 128;
  const long xrow = ((long)b * N_ + row0 + wave * 16 + m) * D_;

  v8f acc[8];
#pragma unroll
  for (int i = 0; i < 8; ++i) acc[i] = vzero8();

  // ---- h = x @ W_down[e] ----
  for (int kc = 0; kc < D_ / 32; ++kc) {
    const v16bf a = load_bf16_frag(xb + xrow + kc * 32, g);
#pragma unroll
    for (int nt = 0; nt < 8; ++nt) {
      const v16bf bf = *(const v16bf*)(wdT + ((long)e * BTL_ + nt * 16 + nlo) * D_ +
                                       kc * 32 + 16 * g);
      acc[nt] = wmma_bf16(a, bf, acc[nt]);
    }
  }
  // ---- h (C layout) -> hfrag (A-frag-native), wave-private ----
#pragma unroll
  for (int nt = 0; nt < 8; ++nt)
#pragma unroll
    for (int vv = 0; vv < 8; ++vv) {
      const int mm = vv + 8 * g;
      const int c = nt * 16 + nlo;
      const int kc2 = c >> 5, kk = c & 31;
      hfrag[(((wave << 2) + kc2) * 32 + afrag_lane(mm, kk)) * 16 + afrag_slot(kk)] =
          (__bf16)acc[nt][vv];
    }

  // ---- q/k/v = h @ W ----
  for (int wsel = 0; wsel < 3; ++wsel) {
    const __bf16* Wt = ((wsel == 0) ? wqT : (wsel == 1) ? wkT : wvT) + (long)e * BTL_ * BTL_;
    v8f acc2[8];
#pragma unroll
    for (int i = 0; i < 8; ++i) acc2[i] = vzero8();
#pragma unroll
    for (int kc = 0; kc < 4; ++kc) {
      const v16bf a = *(const v16bf*)(hfrag + (((wave << 2) + kc) * 32 + lane) * 16);
#pragma unroll
      for (int nt = 0; nt < 8; ++nt) {
        const v16bf bf = *(const v16bf*)(Wt + (nt * 16 + nlo) * BTL_ + kc * 32 + 16 * g);
        acc2[nt] = wmma_bf16(a, bf, acc2[nt]);
      }
    }
    if (wsel < 2) {
      __bf16* outp = (wsel == 0) ? qo : ko;
#pragma unroll
      for (int nt = 0; nt < 8; ++nt)
#pragma unroll
        for (int vv = 0; vv < 8; ++vv)
          outp[((long)b * N_ + row0 + wave * 16 + vv + 8 * g) * BTL_ + nt * 16 + nlo] =
              (__bf16)acc2[nt][vv];
    } else {
      // vT[b][c][token] so PV B-fragments are K(=token)-contiguous
#pragma unroll
      for (int nt = 0; nt < 8; ++nt)
#pragma unroll
        for (int vv = 0; vv < 8; ++vv)
          vT[((long)b * BTL_ + nt * 16 + nlo) * N_ + row0 + wave * 16 + vv + 8 * g] =
              (__bf16)acc2[nt][vv];
    }
  }
}

// ------------------------------------------------------------------ attention + W_up + combine
__global__ __launch_bounds__(256) void k_attn(
    const __bf16* __restrict__ q, const __bf16* __restrict__ k, const __bf16* __restrict__ vT,
    const float* __restrict__ gatew, const int* __restrict__ pos,
    const __bf16* __restrict__ wupT, float* __restrict__ out, int e) {
  __shared__ __align__(32) __bf16 pfrag[8 * 2 * 32 * 16];  // 16 KB [wave][kc][lane][16]
  __shared__ __align__(32) __bf16 ofrag[8 * 4 * 32 * 16];  // 32 KB [wave][kc][lane][16]
  __shared__ float kmsk[64];
  __shared__ int kposs[64];

  const int tid = threadIdx.x, lane = tid & 31, wave = tid >> 5;
  const int g = lane >> 4, nlo = lane & 15, m = lane & 15;
  const int bb = blockIdx.y, qt = blockIdx.x, row0 = qt * 128;
  const long tokbase = (long)bb * N_;
  const float scale = 0.08838834764831845f;  // 1/sqrt(128)

  // Q A-fragments: two contiguous 16B bf16 runs per lane, direct from global
  v16bf qf[4];
#pragma unroll
  for (int kc = 0; kc < 4; ++kc)
    qf[kc] = load_bf16_frag(q + (tokbase + row0 + wave * 16 + m) * BTL_ + kc * 32, g);

  int qpos[8];
#pragma unroll
  for (int vv = 0; vv < 8; ++vv)
    qpos[vv] = pos[tokbase + row0 + wave * 16 + vv + 8 * g];

  float mrow[8], lrow[8];
#pragma unroll
  for (int vv = 0; vv < 8; ++vv) { mrow[vv] = -1e30f; lrow[vv] = 0.f; }
  v8f oacc[8];
#pragma unroll
  for (int i = 0; i < 8; ++i) oacc[i] = vzero8();

  const int nkt = 2 * (qt + 1);
  for (int kt = 0; kt < nkt; ++kt) {
    const int kbase = kt * 64;
    if (tid < 64) {
      kmsk[tid] = (gatew[(tokbase + kbase + tid) * E_ + e] > 0.f) ? 1.f : 0.f;
      kposs[tid] = pos[tokbase + kbase + tid];
    }
    // prefetch next tile's K rows and vT rows into the cache hierarchy
    if (kt + 1 < nkt) {
      const int nxt = kbase + 64;
      __builtin_prefetch(k + (tokbase + nxt + lane * 2) * BTL_, 0, 1);
      __builtin_prefetch(vT + ((long)bb * BTL_ + lane * 4) * N_ + nxt, 0, 1);
    }
    __syncthreads();
    // ---- S = Q K^T : B-fragments direct from row-major K (K-contiguous along c) ----
    v8f sacc[4];
#pragma unroll
    for (int i = 0; i < 4; ++i) sacc[i] = vzero8();
#pragma unroll
    for (int kc = 0; kc < 4; ++kc)
#pragma unroll
      for (int nt = 0; nt < 4; ++nt) {
        const v16bf bf = *(const v16bf*)(k + (tokbase + kbase + nt * 16 + nlo) * BTL_ +
                                         kc * 32 + 16 * g);
        sacc[nt] = wmma_bf16(qf[kc], bf, sacc[nt]);
      }
    // hoist per-lane key mask/pos (invariant over vv)
    float km[4]; int kp[4];
#pragma unroll
    for (int nt = 0; nt < 4; ++nt) {
      km[nt] = kmsk[nt * 16 + nlo];
      kp[nt] = kposs[nt * 16 + nlo];
    }
    // ---- online softmax on C-layout strip ----
#pragma unroll
    for (int vv = 0; vv < 8; ++vv) {
      float m0 = -1e30f;
#pragma unroll
      for (int nt = 0; nt < 4; ++nt) {
        float s = sacc[nt][vv] * scale;
        const bool valid = (km[nt] > 0.f) && (qpos[vv] >= kp[nt]);
        s = valid ? s : -1e30f;
        sacc[nt][vv] = s;
        m0 = fmaxf(m0, s);
      }
#pragma unroll
      for (int off = 1; off < 16; off <<= 1) m0 = fmaxf(m0, __shfl_xor(m0, off, 16));
      const float mnew = fmaxf(mrow[vv], m0);
      const float f = __expf(mrow[vv] - mnew);
      mrow[vv] = mnew;
      float psum = 0.f;
#pragma unroll
      for (int nt = 0; nt < 4; ++nt) {
        const float p = __expf(sacc[nt][vv] - mnew);
        sacc[nt][vv] = p;
        psum += p;
      }
#pragma unroll
      for (int off = 1; off < 16; off <<= 1) psum += __shfl_xor(psum, off, 16);
      lrow[vv] = lrow[vv] * f + psum;
#pragma unroll
      for (int nt = 0; nt < 8; ++nt) oacc[nt][vv] *= f;
    }
    // ---- P (C layout) -> pfrag (A-frag-native), wave-private ----
#pragma unroll
    for (int nt = 0; nt < 4; ++nt)
#pragma unroll
      for (int vv = 0; vv < 8; ++vv) {
        const int mm = vv + 8 * g;
        const int c = nt * 16 + nlo;
        const int kc2 = c >> 5, kk = c & 31;
        pfrag[(((wave << 1) + kc2) * 32 + afrag_lane(mm, kk)) * 16 + afrag_slot(kk)] =
            (__bf16)sacc[nt][vv];
      }
    // ---- O += P @ V : B-fragments direct from vT (K-contiguous along token) ----
#pragma unroll
    for (int kc = 0; kc < 2; ++kc) {
      const v16bf a = *(const v16bf*)(pfrag + (((wave << 1) + kc) * 32 + lane) * 16);
#pragma unroll
      for (int nt = 0; nt < 8; ++nt) {
        const v16bf bf = *(const v16bf*)(vT + ((long)bb * BTL_ + nt * 16 + nlo) * N_ +
                                         kbase + kc * 32 + 16 * g);
        oacc[nt] = wmma_bf16(a, bf, oacc[nt]);
      }
    }
    __syncthreads();  // protect kmsk/kposs for next iteration
  }

  // ---- epilogue: scale by gate/l, relayout O, Y = O @ W_up[e], out += Y ----
  float sc[8];
#pragma unroll
  for (int vv = 0; vv < 8; ++vv) {
    const float gate = gatew[(tokbase + row0 + wave * 16 + vv + 8 * g) * E_ + e];
    sc[vv] = (lrow[vv] > 0.f) ? gate / lrow[vv] : 0.f;
  }
#pragma unroll
  for (int nt = 0; nt < 8; ++nt)
#pragma unroll
    for (int vv = 0; vv < 8; ++vv) {
      const int mm = vv + 8 * g;
      const int c = nt * 16 + nlo;
      const int kc2 = c >> 5, kk = c & 31;
      ofrag[(((wave << 2) + kc2) * 32 + afrag_lane(mm, kk)) * 16 + afrag_slot(kk)] =
          (__bf16)(oacc[nt][vv] * sc[vv]);
    }
  v16bf oa[4];
#pragma unroll
  for (int kc = 0; kc < 4; ++kc)
    oa[kc] = *(const v16bf*)(ofrag + (((wave << 2) + kc) * 32 + lane) * 16);

  for (int nb = 0; nb < D_ / 128; ++nb) {
    v8f yacc[8];
#pragma unroll
    for (int i = 0; i < 8; ++i) yacc[i] = vzero8();
#pragma unroll
    for (int kc = 0; kc < 4; ++kc)
#pragma unroll
      for (int nt = 0; nt < 8; ++nt) {
        const v16bf bf = *(const v16bf*)(wupT + ((long)e * D_ + nb * 128 + nt * 16 + nlo) * BTL_ +
                                         kc * 32 + 16 * g);
        yacc[nt] = wmma_bf16(oa[kc], bf, yacc[nt]);
      }
#pragma unroll
    for (int nt = 0; nt < 8; ++nt)
#pragma unroll
      for (int vv = 0; vv < 8; ++vv) {
        const long idx = (tokbase + row0 + wave * 16 + vv + 8 * g) * (long)D_ +
                         nb * 128 + nt * 16 + nlo;
        out[idx] += yacc[nt][vv];  // experts serialized on stream: race-free
      }
  }
}

// ------------------------------------------------------------------ launch
extern "C" void kernel_launch(void* const* d_in, const int* in_sizes, int n_in,
                              void* d_out, int out_size, void* d_ws, size_t ws_size,
                              hipStream_t stream) {
  const float* x   = (const float*)d_in[0];
  const int* pos   = (const int*)d_in[1];
  const unsigned char* act = (const unsigned char*)d_in[2];
  const float* wg  = (const float*)d_in[3];
  const float* wd  = (const float*)d_in[4];
  const float* wq  = (const float*)d_in[5];
  const float* wk  = (const float*)d_in[6];
  const float* wv  = (const float*)d_in[7];
  const float* wup = (const float*)d_in[8];
  const int* topk  = (const int*)d_in[9];
  float* out = (float*)d_out;

  char* ws = (char*)d_ws;
  float*  gatew = (float*)ws;                              // 256 KB
  float*  red   = (float*)(ws + (512 << 10));              // 32 f32
  __bf16* wdT   = (__bf16*)(ws + (1 << 20));               // 2 MB
  __bf16* wqT   = (__bf16*)(ws + (3 << 20));               // 256 KB
  __bf16* wkT   = (__bf16*)(ws + (3 << 20) + (512 << 10)); // 256 KB
  __bf16* wvT   = (__bf16*)(ws + (4 << 20));               // 256 KB
  __bf16* wupT  = (__bf16*)(ws + (4 << 20) + (512 << 10)); // 2 MB
  __bf16* qb    = (__bf16*)(ws + (7 << 20));               // 2 MB
  __bf16* kb    = (__bf16*)(ws + (9 << 20));               // 2 MB
  __bf16* vTb   = (__bf16*)(ws + (11 << 20));              // 2 MB
  __bf16* xbb   = (__bf16*)(ws + (13 << 20));              // 16 MB (29 MB total)

  k_init<<<2048, 256, 0, stream>>>(out, red);
  k_prep<<<2048, 256, 0, stream>>>(x, wd, wq, wk, wv, wup, xbb, wdT, wqT, wkT, wvT, wupT);
  k_router<<<(B_ * N_ + 255) / 256, 256, 0, stream>>>(x, act, wg, gatew, red);
  k_loss<<<1, 32, 0, stream>>>(red, topk, out);
  for (int e = 0; e < E_; ++e) {
    k_qkv<<<dim3(N_ / 128, B_), 256, 0, stream>>>(xbb, wdT, wqT, wkT, wvT, qb, kb, vTb, e);
    k_attn<<<dim3(N_ / 128, B_), 256, 0, stream>>>(qb, kb, vTb, gatew, pos, wupT, out, e);
  }
}
